// DSFS_90872918049360
// MI455X (gfx1250) — matrix operand order repaired and checked by reading, not compiled, not run on hardware
//
#include <hip/hip_runtime.h>

typedef __attribute__((ext_vector_type(16))) _Float16 v16h;
typedef __attribute__((ext_vector_type(8)))  _Float16 v8h;
typedef __attribute__((ext_vector_type(4)))  _Float16 v4h;
typedef __attribute__((ext_vector_type(8)))  float    v8f;
typedef __attribute__((ext_vector_type(4)))  float    v4f;

// Fused GEMM: C = relu(A @ B + bias) (+ Res), A:[M,K] f32, B:[K,N] f32.
// f32 -> f16 fused into LDS staging; f32 accumulate via v_wmma_f32_16x16x32_f16.
// Block tile 128 x BN (BN = 64 or 128), BK = 32, double-buffered LDS.
// 256 threads = 8 waves (wave32) arranged 4x2; each wave owns 32 x (BN/2)
// = 2 x (BN/32) WMMA 16x16 tiles. blockIdx.z batches independent GEMMs.
template<int BN>
__global__ __launch_bounds__(256)
void gemm_bias_relu_wmma(const float* __restrict__ A, int lda, long long strideA,
                         const float* __restrict__ B, int ldb, long long strideB,
                         const float* __restrict__ bias, long long strideBias,
                         const float* __restrict__ Res, int ldres,
                         float* __restrict__ C, int ldc, long long strideC,
                         int M, int N, int K)
{
    constexpr int NT    = BN / 32;   // 16-wide WMMA tiles per wave in N (2 or 4)
    constexpr int LDS_K = 40;        // padded K stride: 80B rows -> 16B aligned,
                                     // 20-bank stride -> conflict-free b128 reads
    __shared__ __align__(16) _Float16 As[2][128][LDS_K];
    __shared__ __align__(16) _Float16 Bs[2][BN][LDS_K];   // transposed: Bs[n][k]

    const long long z = blockIdx.z;
    A    += z * strideA;
    B    += z * strideB;
    bias += z * strideBias;
    C    += z * strideC;

    const int tid  = threadIdx.x;
    const int lane = tid & 31;
    const int wid  = tid >> 5;
    const int wm   = wid & 3;             // m offset wm*32
    const int wn   = wid >> 2;            // n offset wn*(BN/2)
    const int bm   = blockIdx.y * 128;
    const int bn   = blockIdx.x * BN;

    const int l16    = lane & 15;
    const int hi16   = lane >> 4;
    const int akhalf = hi16 << 3;         // A frag: K base 0/8  (pairs with +16)
    const int bkbase = hi16 << 3;         // B frag: K base 0/8  (pairs with +16)

    v8f acc[2][NT] = {};

    // ---- stage one 128x32 A tile + 32xBN B tile (f32 -> f16, B transposed) ----
    auto stage = [&](int buf, int k0) {
        #pragma unroll
        for (int it = 0; it < 4; ++it) {
            int idx = tid + it * 256;                 // 0..1023 float4s
            int r   = idx >> 3;                       // 0..127
            int c4  = (idx & 7) << 2;                 // 0,4,...,28
            v4f av = *(const v4f*)(A + (long long)(bm + r) * lda + k0 + c4);
            v4h ah = { (_Float16)av.x, (_Float16)av.y, (_Float16)av.z, (_Float16)av.w };
            *(v4h*)&As[buf][r][c4] = ah;
        }
        #pragma unroll
        for (int it = 0; it < BN / 32; ++it) {
            int idx = tid + it * 256;                 // over 8*BN float4s
            int r   = idx / (BN / 4);                 // k row 0..31
            int c4  = (idx % (BN / 4)) << 2;          // 0..BN-4
            v4f bv = *(const v4f*)(B + (long long)(k0 + r) * ldb + bn + c4);
            Bs[buf][c4 + 0][r] = (_Float16)bv.x;
            Bs[buf][c4 + 1][r] = (_Float16)bv.y;
            Bs[buf][c4 + 2][r] = (_Float16)bv.z;
            Bs[buf][c4 + 3][r] = (_Float16)bv.w;
        }
    };

    // ---- fragments per ISA 16-bit layouts + 2xNT WMMAs ----
    auto compute = [&](int buf) {
        v16h af[2], bfr[NT];
        #pragma unroll
        for (int i = 0; i < 2; ++i) {
            int row = wm * 32 + i * 16 + l16;
            v8h lo = *(const v8h*)&As[buf][row][akhalf];       // K 0-7  / 8-15
            v8h hi = *(const v8h*)&As[buf][row][akhalf + 16];  // K 16-23/ 24-31
            af[i] = __builtin_shufflevector(lo, hi, 0,1,2,3,4,5,6,7,8,9,10,11,12,13,14,15);
        }
        #pragma unroll
        for (int j = 0; j < NT; ++j) {
            int col = wn * (BN / 2) + j * 16 + l16;
            v8h lo = *(const v8h*)&Bs[buf][col][bkbase];       // K 0-7  / 8-15
            v8h hi = *(const v8h*)&Bs[buf][col][bkbase + 16];  // K 16-23/ 24-31
            bfr[j] = __builtin_shufflevector(lo, hi, 0,1,2,3,4,5,6,7,8,9,10,11,12,13,14,15);
        }
        #pragma unroll
        for (int i = 0; i < 2; ++i)
            #pragma unroll
            for (int j = 0; j < NT; ++j)
                acc[i][j] = __builtin_amdgcn_wmma_f32_16x16x32_f16(
                    /*neg_a=*/false, af[i], /*neg_b=*/false, bfr[j],
                    /*c_mod=*/(short)0, acc[i][j],
                    /*reuse_a=*/false, /*reuse_b=*/false);
    };

    // ---- double-buffered main loop: one barrier per k-step ----
    const int nk = K >> 5;
    stage(0, 0);
    __syncthreads();
    for (int ks = 0; ks < nk - 1; ++ks) {
        stage((ks + 1) & 1, (ks + 1) << 5);   // prefetch next tile into other buffer
        compute(ks & 1);
        __syncthreads();
    }
    compute((nk - 1) & 1);

    // ---- epilogue: bias + relu (+ residual), per 16x16 f32 C/D layout ----
    const int rofs = hi16 << 3;               // lanes 16-31 hold M = r+8
    #pragma unroll
    for (int i = 0; i < 2; ++i) {
        #pragma unroll
        for (int j = 0; j < NT; ++j) {
            int colg = bn + wn * (BN / 2) + j * 16 + l16;
            float bv = bias[colg];
            #pragma unroll
            for (int r = 0; r < 8; ++r) {
                int rowg = bm + wm * 32 + i * 16 + r + rofs;
                float v = acc[i][j][r] + bv;
                v = v > 0.0f ? v : 0.0f;
                if (Res) v += Res[(long long)rowg * ldres + colg];
                C[(long long)rowg * ldc + colg] = v;
            }
        }
    }
}

extern "C" void kernel_launch(void* const* d_in, const int* in_sizes, int n_in,
                              void* d_out, int out_size, void* d_ws, size_t ws_size,
                              hipStream_t stream) {
    const float* x       = (const float*)d_in[0]; // [4096,32,64]
    const float* feat_w  = (const float*)d_in[1]; // [32,64,64]
    const float* feat_b  = (const float*)d_in[2]; // [32,64]
    const float* trans_w = (const float*)d_in[3]; // [2048,2048]
    const float* trans_b = (const float*)d_in[4]; // [2048]
    const float* tw_w    = (const float*)d_in[5]; // [2048,8192]
    const float* tw_b    = (const float*)d_in[6]; // [8192]
    const float* tb_w    = (const float*)d_in[7]; // [2048,128]
    const float* tb_b    = (const float*)d_in[8]; // [128]
    float* out = (float*)d_out;

    const int B = 4096, FH = 2048, D0D1 = 64 * 128, D1 = 128;
    float* hf  = (float*)d_ws;                    // [4096,2048] f32
    float* res = hf + (size_t)B * FH;             // [4096,2048] f32

    dim3 blk(256);

    // 1) per-feature MLP: hf[b, f*64+h] = relu(x[b,f,:] @ feat_w[f] + feat_b[f])
    //    batched over blockIdx.z = f (32 GEMMs of 4096x64x64)
    gemm_bias_relu_wmma<64><<<dim3(1, 32, 32), blk, 0, stream>>>(
        x, FH, 64,            // A = x[:, f, :]
        feat_w, 64, 4096,     // B = feat_w[f]
        feat_b, 64,           // bias = feat_b[f]
        nullptr, 0,
        hf, FH, 64,           // C column block f*64
        B, 64, 64);

    // 2) res = relu(hf @ trans_w + trans_b) + hf   (4096x2048x2048)
    gemm_bias_relu_wmma<128><<<dim3(FH / 128, B / 128, 1), blk, 0, stream>>>(
        hf, FH, 0, trans_w, FH, 0, trans_b, 0,
        hf, FH,               // residual
        res, FH, 0, B, FH, FH);

    // 3) weight = relu(res @ tw_w + tw_b)          (4096x2048x8192) -> out
    gemm_bias_relu_wmma<128><<<dim3(D0D1 / 128, B / 128, 1), blk, 0, stream>>>(
        res, FH, 0, tw_w, D0D1, 0, tw_b, 0,
        nullptr, 0,
        out, D0D1, 0, B, D0D1, FH);

    // 4) bias = relu(res @ tb_w + tb_b)            (4096x2048x128) -> out tail
    gemm_bias_relu_wmma<64><<<dim3(D1 / 64, B / 128, 1), blk, 0, stream>>>(
        res, FH, 0, tb_w, D1, 0, tb_b, 0,
        nullptr, 0,
        out + (size_t)B * D0D1, D1, 0, B, D1, FH);
}
